// Autoencoder_53987738911503
// MI455X (gfx1250) — compile-verified, matrix operand order checked by reading, not run on hardware
//
#include <hip/hip_runtime.h>
#include <hip/hip_bf16.h>
#include <cstdint>
#include <cstddef>

typedef __bf16 bf16_t;
typedef __attribute__((ext_vector_type(16))) __bf16 v16bf;
typedef __attribute__((ext_vector_type(8)))  __bf16 v8bf;
typedef __attribute__((ext_vector_type(8)))  float  v8f;

// ---------------------------------------------------------------------------
// WMMA fragment helper: halves 0..7 <- K = kb+koff..+7, halves 8..15 <-
// K = kb+16+koff..+7 (CDNA5 16-bit A/B layout, wave32).
// ---------------------------------------------------------------------------
__device__ __forceinline__ v16bf frag16(const v8bf* row, int kb, int koff) {
  const v8bf lo = row[(kb + koff) >> 3];
  const v8bf hi = row[(kb + 16 + koff) >> 3];
  return __builtin_shufflevector(lo, hi,
      0, 1, 2, 3, 4, 5, 6, 7, 8, 9, 10, 11, 12, 13, 14, 15);
}

// ---------------------------------------------------------------------------
// conv1 only (CIN=1): direct fp32 form — K=9 would be 72% padding under WMMA.
// ---------------------------------------------------------------------------
template<int CIN, int COUT, int HIN, int WIN>
__global__ __launch_bounds__(256) void conv3x3_s2_relu(
    const float* __restrict__ x, const float* __restrict__ w,
    const float* __restrict__ bias, float* __restrict__ y)
{
  constexpr int HOUT = HIN / 2, WOUT = WIN / 2;
  const int n  = blockIdx.z;
  const int co = blockIdx.y;
  __shared__ float wsh[CIN * 9];
  __shared__ float bsh;
  for (int i = threadIdx.x; i < CIN * 9; i += blockDim.x)
    wsh[i] = w[(size_t)co * CIN * 9 + i];
  if (threadIdx.x == 0) bsh = bias[co];
  __syncthreads();

  const int p = blockIdx.x * blockDim.x + threadIdx.x;
  if (p >= HOUT * WOUT) return;
  const int oy = p / WOUT, ox = p % WOUT;

  const float* __restrict__ xb = x + (size_t)n * CIN * HIN * WIN;
  float acc = bsh;
  for (int ci = 0; ci < CIN; ++ci) {
    const float* __restrict__ xc = xb + (size_t)ci * HIN * WIN;
    const float* __restrict__ wc = &wsh[ci * 9];
    #pragma unroll
    for (int ky = 0; ky < 3; ++ky) {
      const int iy = oy * 2 + ky - 1;
      if (iy < 0 || iy >= HIN) continue;
      #pragma unroll
      for (int kx = 0; kx < 3; ++kx) {
        const int ix = ox * 2 + kx - 1;
        if (ix < 0 || ix >= WIN) continue;
        acc = fmaf(xc[(size_t)iy * WIN + ix], wc[ky * 3 + kx], acc);
      }
    }
  }
  y[(((size_t)n * COUT + co) * HOUT + oy) * WOUT + ox] = fmaxf(acc, 0.0f);
}

// ---------------------------------------------------------------------------
// Weight pack: OIHW fp32 -> bf16 [COUT][KPAD] (N-major, K zero-padded to a
// multiple of 32) so conv B-fragments are two contiguous 16B loads.
// ---------------------------------------------------------------------------
template<int CIN, int COUT>
__global__ __launch_bounds__(256) void pack_weights_bf16(
    const float* __restrict__ w, bf16_t* __restrict__ wt)
{
  constexpr int K = CIN * 9, KPAD = (K + 31) & ~31;
  const int idx = blockIdx.x * 256 + threadIdx.x;
  if (idx >= COUT * KPAD) return;
  const int co = idx / KPAD, k = idx % KPAD;
  wt[idx] = (bf16_t)((k < K) ? w[(size_t)co * K + k] : 0.0f);
}

// ---------------------------------------------------------------------------
// Implicit-GEMM conv (k=3, s=2, p=1) + ReLU on the matrix pipe:
//   M = HOUT*WOUT pixels (chunked, zero-padded), K = CIN*9 (padded to 32),
//   N = COUT. A = im2col chunk in LDS (bf16); B = packed bf16 weights from
//   global (L1/L2 resident). v_wmma_f32_16x16x32_bf16, f32 accumulate,
//   bias + ReLU fused into the store.
// ---------------------------------------------------------------------------
template<int CIN, int COUT, int HIN, int WIN, int MCHUNK>
__global__ __launch_bounds__(256) void conv3x3_s2_relu_wmma(
    const float* __restrict__ x,        // [N][CIN][HIN][WIN]
    const bf16_t* __restrict__ wt,      // [COUT][KPAD] packed weights
    const float* __restrict__ bias,     // [COUT]
    float* __restrict__ y)              // [N][COUT][HOUT][WOUT]
{
  constexpr int HOUT = HIN / 2, WOUT = WIN / 2;
  constexpr int M    = HOUT * WOUT;
  constexpr int K    = CIN * 9;
  constexpr int KPAD = (K + 31) & ~31;
  constexpr int MT   = MCHUNK / 16;
  constexpr int NT   = COUT / 16;

  __shared__ __align__(16) bf16_t As[MCHUNK][KPAD];

  const int n    = blockIdx.y;
  const int p0   = blockIdx.x * MCHUNK;
  const int tid  = threadIdx.x;
  const int lane = tid & 31;
  const int wave = tid >> 5;
  const int mloc = lane & 15;
  const int koff = (lane >> 4) * 8;
  const int crow = (lane >> 4) * 8;

  // im2col chunk -> LDS (bf16), zero-padding K tail, M tail, and halo.
  const float* __restrict__ xb = x + (size_t)n * CIN * HIN * WIN;
  for (int idx = tid; idx < MCHUNK * KPAD; idx += 256) {
    const int r = idx / KPAD;
    const int k = idx % KPAD;
    float v = 0.0f;
    const int p = p0 + r;
    if (p < M && k < K) {
      const int ci = k / 9,   t  = k - ci * 9;
      const int ky = t / 3,   kx = t - ky * 3;
      const int oy = p / WOUT, ox = p - oy * WOUT;
      const int iy = oy * 2 + ky - 1, ix = ox * 2 + kx - 1;
      if (iy >= 0 && iy < HIN && ix >= 0 && ix < WIN)
        v = xb[(size_t)ci * HIN * WIN + (size_t)iy * WIN + ix];
    }
    As[r][k] = (bf16_t)v;
  }
  __syncthreads();

  // Each wave sweeps its (mtile, ntile) tasks.
  for (int task = wave; task < MT * NT; task += 8) {
    const int mt = task / NT, nt = task % NT;
    const int co = nt * 16 + mloc;
    const v8bf* __restrict__ arow = (const v8bf*)(&As[mt * 16 + mloc][0]);
    const v8bf* __restrict__ brow = (const v8bf*)(wt + (size_t)co * KPAD);

    v8f acc = {};
    #pragma unroll
    for (int kb = 0; kb < KPAD; kb += 32) {
      const v16bf afrag = frag16(arow, kb, koff);
      const v16bf bfrag = frag16(brow, kb, koff);
      acc = __builtin_amdgcn_wmma_f32_16x16x32_bf16(
          false, afrag, false, bfrag, (short)0, acc, false, false);
    }

    const float bb = bias[co];
    float* __restrict__ yc = y + ((size_t)n * COUT + co) * M;
    #pragma unroll
    for (int r = 0; r < 8; ++r) {
      const int p = p0 + mt * 16 + crow + r;
      if (p < M) yc[p] = fmaxf(acc[r] + bb, 0.0f);
    }
  }
}

// ---------------------------------------------------------------------------
// Decoder: ConvTranspose2d(k=3, s=2, p=1, output_padding=OP) as a gather over
// the stride-2 input-dilated image, pad_lo = 1.  out = 2*HIN - 1 + OP.
// Activation: ReLU, or sigmoid on the final layer.
// ---------------------------------------------------------------------------
template<int CIN, int COUT, int HIN, int WIN, int OP, bool SIG>
__global__ __launch_bounds__(256) void convt3x3_s2_act(
    const float* __restrict__ x, const float* __restrict__ w,
    const float* __restrict__ bias, float* __restrict__ y)
{
  constexpr int HOUT = 2 * HIN - 1 + OP, WOUT = 2 * WIN - 1 + OP;
  const int n  = blockIdx.z;
  const int co = blockIdx.y;
  __shared__ float wsh[CIN * 9];
  __shared__ float bsh;
  for (int i = threadIdx.x; i < CIN * 9; i += blockDim.x)
    wsh[i] = w[(size_t)co * CIN * 9 + i];
  if (threadIdx.x == 0) bsh = bias[co];
  __syncthreads();

  const int p = blockIdx.x * blockDim.x + threadIdx.x;
  if (p >= HOUT * WOUT) return;
  const int oy = p / WOUT, ox = p % WOUT;

  int iy[3], ix[3];
  bool vy[3], vx[3];
  #pragma unroll
  for (int k = 0; k < 3; ++k) {
    const int ty = oy + k - 1;
    vy[k] = (ty >= 0) && (ty <= 2 * HIN - 2) && ((ty & 1) == 0);
    iy[k] = ty >> 1;
    const int tx = ox + k - 1;
    vx[k] = (tx >= 0) && (tx <= 2 * WIN - 2) && ((tx & 1) == 0);
    ix[k] = tx >> 1;
  }

  float acc = bsh;
  for (int ci = 0; ci < CIN; ++ci) {
    const float* __restrict__ xc = x + ((size_t)n * CIN + ci) * HIN * WIN;
    const float* __restrict__ wc = &wsh[ci * 9];
    #pragma unroll
    for (int ky = 0; ky < 3; ++ky) {
      if (!vy[ky]) continue;
      #pragma unroll
      for (int kx = 0; kx < 3; ++kx) {
        if (!vx[kx]) continue;
        acc = fmaf(xc[(size_t)iy[ky] * WIN + ix[kx]], wc[ky * 3 + kx], acc);
      }
    }
  }
  const float r = SIG ? (1.0f / (1.0f + __expf(-acc))) : fmaxf(acc, 0.0f);
  y[(((size_t)n * COUT + co) * HOUT + oy) * WOUT + ox] = r;
}

// ---------------------------------------------------------------------------
// Tensor Data Mover support (gfx1250). Builds a D# for a contiguous 1-row
// tile of `nelem` fp32 elements and issues TENSOR_LOAD_TO_LDS (6-arg builtin
// on this toolchain). Bit layout per CDNA5 ISA ch.8 (D# groups 0/1).
// ---------------------------------------------------------------------------
#if defined(__gfx1250__) && __has_builtin(__builtin_amdgcn_tensor_load_to_lds)
#define EXPERT_HAVE_TDM 1

typedef __attribute__((ext_vector_type(4))) unsigned int u32x4;
typedef __attribute__((ext_vector_type(8))) int          i32x8;
typedef __attribute__((ext_vector_type(4))) int          i32x4;

__device__ __forceinline__ uint32_t lds_byte_off(const void* p) {
  // generic pointers to LDS carry the workgroup-relative byte offset in the
  // low 32 bits (shared aperture in the high bits)
  return (uint32_t)(uintptr_t)p;
}

__device__ __forceinline__ void tdm_load_1d_f32(const float* gsrc,
                                                uint32_t lds_addr,
                                                uint32_t nelem /* <= 4096 */) {
  const uint64_t ga = (uint64_t)(uintptr_t)gsrc;
  u32x4 g0;
  g0[0] = 1u;                                           // count=1, user D#
  g0[1] = lds_addr;                                     // lds_addr[31:0]
  g0[2] = (uint32_t)ga;                                 // global_addr[31:0]
  g0[3] = (uint32_t)((ga >> 32) & 0x1FFFFFFu)           // global_addr[56:32]
        | (2u << 30);                                   // type = 2 ("image")
  i32x8 g1;
  g1[0] = (int)(2u << 16);                              // data_size=4B; mask=0
  g1[1] = (int)((nelem & 0xFFFFu) << 16);               // tensor_dim0[15:0]
  g1[2] = (int)((nelem >> 16) & 0xFFFFu) | (1 << 16);   // td0[31:16]; td1=1
  g1[3] = (int)((nelem & 0xFFFFu) << 16);               // tile_dim0 = nelem
  g1[4] = 1;                                            // tile_dim1=1; td2=0
  g1[5] = (int)nelem;                                   // td0_stride[31:0]
  g1[6] = 0;                                            // td0_stride[47:32]
  g1[7] = 0;                                            // td1_stride
  const i32x4 z4 = {0, 0, 0, 0};
  const i32x8 z8 = {0, 0, 0, 0, 0, 0, 0, 0};
  __builtin_amdgcn_tensor_load_to_lds(g0, g1, z4, z4, z8, 0);
}
#endif

// ---------------------------------------------------------------------------
// Expert MLP: 64 independent 3-layer MLPs (49->64->64->49) over B=512 rows.
// One block per (expert, 128-row batch chunk); 8 wave32 waves. All three
// layers fused; activations ping-pong in LDS as bf16; GEMMs are
// v_wmma_f32_16x16x32_bf16 (f32 accumulate). Weights are DMA'd into LDS by
// the Tensor Data Mover (wave 0 issues; s_wait_tensorcnt + barrier publish),
// then transposed/bf16-converted in LDS. The next layer's weight DMA is
// issued before the current layer's WMMA loop so it overlaps compute.
// ---------------------------------------------------------------------------
__global__ __launch_bounds__(256) void expert_mlp_wmma(
    const float* __restrict__ h4,                         // [512][64][49]
    const float* __restrict__ W1, const float* __restrict__ b1,
    const float* __restrict__ W2, const float* __restrict__ b2,
    const float* __restrict__ W3, const float* __restrict__ b3,
    float* __restrict__ f3)                               // [512][64][49]
{
  __shared__ __align__(16) bf16_t Xs[128][64];
  __shared__ __align__(16) bf16_t Hs[128][64];
  __shared__ __align__(16) bf16_t Ws[64][64];     // transposed: Ws[n][k], bf16
  __shared__ __align__(16) float  Wstage[64 * 64]; // raw fp32 from TDM
  __shared__ float bs[64];

  const int e    = blockIdx.x;            // expert 0..63
  const int b0   = blockIdx.y * 128;      // batch chunk
  const int tid  = threadIdx.x;
  const int lane = tid & 31;
  const int wave = tid >> 5;
  const int r0   = wave * 16;             // this wave's 16-row M tile
  const int m    = lane & 15;
  const int koff = (lane >> 4) * 8;       // A/B fragment K sub-offset
  const int crow = (lane >> 4) * 8;       // C/D fragment M sub-offset

  const float* Wg[3] = { W1 + (size_t)e * 49 * 64,
                         W2 + (size_t)e * 64 * 64,
                         W3 + (size_t)e * 64 * 49 };
  const float* bg[3] = { b1 + (size_t)e * 64,
                         b2 + (size_t)e * 64,
                         b3 + (size_t)e * 49 };
  const int Kv[3] = { 49, 64, 64 };
  const int Nv[3] = { 64, 64, 49 };

#if EXPERT_HAVE_TDM
  // Kick off the layer-0 weight DMA immediately; it overlaps X staging.
  if (wave == 0)
    tdm_load_1d_f32(Wg[0], lds_byte_off(Wstage), (uint32_t)(Kv[0] * Nv[0]));
#endif

  // Stage input activations, zero-padding K: 49 -> 64.
  for (int idx = tid; idx < 128 * 64; idx += 256) {
    const int r = idx >> 6, i = idx & 63;
    const float v = (i < 49) ? h4[((size_t)(b0 + r) * 64 + e) * 49 + i] : 0.0f;
    Xs[r][i] = (bf16_t)v;
  }

  bf16_t (*in )[64] = Xs;
  bf16_t (*out)[64] = Hs;

  for (int layer = 0; layer < 3; ++layer) {
    const int kv = Kv[layer], nv = Nv[layer];

#if EXPERT_HAVE_TDM
    if (wave == 0) __builtin_amdgcn_s_wait_tensorcnt(0);  // DMA landed
    __syncthreads();   // publish Wstage; prev outputs visible; Ws reads done
#else
    __syncthreads();   // prev layer's Wstage/Ws readers done
    for (int idx = tid; idx < kv * nv; idx += 256) Wstage[idx] = Wg[layer][idx];
    __syncthreads();
#endif

    // LDS transpose + bf16 convert: Ws[n][k], zero-padded to 64x64.
    for (int idx = tid; idx < 64 * 64; idx += 256) {
      const int k = idx >> 6, nn = idx & 63;
      const float v = (k < kv && nn < nv) ? Wstage[k * nv + nn] : 0.0f;
      Ws[nn][k] = (bf16_t)v;
    }
    if (tid < 64) bs[tid] = (tid < nv) ? bg[layer][tid] : 0.0f;
    __syncthreads();   // Ws ready; Wstage free again

#if EXPERT_HAVE_TDM
    // Overlap the next layer's weight DMA with this layer's WMMA loop.
    if (layer < 2 && wave == 0)
      tdm_load_1d_f32(Wg[layer + 1], lds_byte_off(Wstage),
                      (uint32_t)(Kv[layer + 1] * Nv[layer + 1]));
#endif

    // 16x64 tile per wave: 4 accumulators, K = 64 in two 32-wide WMMA steps.
    v8f acc[4] = {};
    const v8bf* __restrict__ xrow = (const v8bf*)(&in[r0 + m][0]);
    #pragma unroll
    for (int kb = 0; kb < 64; kb += 32) {
      const v16bf afrag = frag16(xrow, kb, koff);
      #pragma unroll
      for (int nt = 0; nt < 4; ++nt) {
        const v8bf* __restrict__ wrow = (const v8bf*)(&Ws[nt * 16 + m][0]);
        const v16bf bfrag = frag16(wrow, kb, koff);
        acc[nt] = __builtin_amdgcn_wmma_f32_16x16x32_bf16(
            false, afrag, false, bfrag, (short)0, acc[nt], false, false);
      }
    }

    // Epilogue: bias + ReLU. Last layer writes fp32 to global (N valid = 49).
    if (layer == 2) {
      #pragma unroll
      for (int nt = 0; nt < 4; ++nt) {
        const int nn = nt * 16 + m;
        if (nn < 49) {
          const float bb = bs[nn];
          #pragma unroll
          for (int r = 0; r < 8; ++r) {
            const float v = fmaxf(acc[nt][r] + bb, 0.0f);
            f3[((size_t)(b0 + r0 + crow + r) * 64 + e) * 49 + nn] = v;
          }
        }
      }
    } else {
      #pragma unroll
      for (int nt = 0; nt < 4; ++nt) {
        const int nn = nt * 16 + m;
        const float bb = bs[nn];
        #pragma unroll
        for (int r = 0; r < 8; ++r) {
          const float v = fmaxf(acc[nt][r] + bb, 0.0f);
          out[r0 + crow + r][nn] = (bf16_t)v;
        }
      }
      bf16_t (*t)[64] = in; in = out; out = t;  // ping-pong
    }
  }
}

// ---------------------------------------------------------------------------
// Host-side orchestration. Workspace:
//   bufA (52 MB): h1 / h3 / f3 / d2      bufB (41 MB): h2 / h4 / d1 / d3
//   packed bf16 conv weights: 3 x 64 KB slots above 93 MB.
// ---------------------------------------------------------------------------
extern "C" void kernel_launch(void* const* d_in, const int* in_sizes, int n_in,
                              void* d_out, int out_size, void* d_ws, size_t ws_size,
                              hipStream_t stream) {
  (void)in_sizes; (void)n_in; (void)out_size; (void)ws_size;
  const float* x   = (const float*)d_in[0];
  const float* we1 = (const float*)d_in[1];  const float* be1 = (const float*)d_in[2];
  const float* we2 = (const float*)d_in[3];  const float* be2 = (const float*)d_in[4];
  const float* we3 = (const float*)d_in[5];  const float* be3 = (const float*)d_in[6];
  const float* we4 = (const float*)d_in[7];  const float* be4 = (const float*)d_in[8];
  const float* W1  = (const float*)d_in[9];  const float* b1  = (const float*)d_in[10];
  const float* W2  = (const float*)d_in[11]; const float* b2  = (const float*)d_in[12];
  const float* W3  = (const float*)d_in[13]; const float* b3  = (const float*)d_in[14];
  const float* wd1 = (const float*)d_in[15]; const float* bd1 = (const float*)d_in[16];
  const float* wd2 = (const float*)d_in[17]; const float* bd2 = (const float*)d_in[18];
  const float* wd3 = (const float*)d_in[19]; const float* bd3 = (const float*)d_in[20];
  const float* wd4 = (const float*)d_in[21]; const float* bd4 = (const float*)d_in[22];
  float* out = (float*)d_out;

  char*   ws   = (char*)d_ws;
  float*  bufA = (float*)ws;                          // 52 MB region
  float*  bufB = (float*)(ws + (52ull << 20));        // 41 MB region
  bf16_t* wt2  = (bf16_t*)(ws + (93ull << 20));                  // 16 x 96
  bf16_t* wt3  = (bf16_t*)(ws + (93ull << 20) + (64u << 10));    // 32 x 160
  bf16_t* wt4  = (bf16_t*)(ws + (93ull << 20) + (128u << 10));   // 64 x 288

  const int N = 512;

  // Pack encoder conv weights -> bf16 [COUT][KPAD] (tiny, runs once per call).
  pack_weights_bf16< 8, 16><<<dim3( 6), 256, 0, stream>>>(we2, wt2);
  pack_weights_bf16<16, 32><<<dim3(20), 256, 0, stream>>>(we3, wt3);
  pack_weights_bf16<32, 64><<<dim3(72), 256, 0, stream>>>(we4, wt4);

  // Encoder: 112 -> 56 (direct), then implicit-GEMM WMMA: 56->28->14->7
  conv3x3_s2_relu<1, 8, 112, 112><<<dim3(13, 8, N), 256, 0, stream>>>(x, we1, be1, bufA);
  conv3x3_s2_relu_wmma< 8, 16,  56,  56, 128><<<dim3(7, N), 256, 0, stream>>>(bufA, wt2, be2, bufB);
  conv3x3_s2_relu_wmma<16, 32,  28,  28, 128><<<dim3(2, N), 256, 0, stream>>>(bufB, wt3, be3, bufA);
  conv3x3_s2_relu_wmma<32, 64,  14,  14,  64><<<dim3(1, N), 256, 0, stream>>>(bufA, wt4, be4, bufB);

  // 64 experts x 3-layer MLP, fused, WMMA bf16 (f32 accumulate) + TDM weights
  expert_mlp_wmma<<<dim3(64, 4), 256, 0, stream>>>(bufB, W1, b1, W2, b2, W3, b3, bufA);

  // Decoder: 7 -> 13 -> 25 -> 50 -> 100
  convt3x3_s2_act<64, 32,  7,  7, 0, false><<<dim3( 1, 32, N), 256, 0, stream>>>(bufA, wd1, bd1, bufB);
  convt3x3_s2_act<32, 16, 13, 13, 0, false><<<dim3( 3, 16, N), 256, 0, stream>>>(bufB, wd2, bd2, bufA);
  convt3x3_s2_act<16,  8, 25, 25, 1, false><<<dim3(10,  8, N), 256, 0, stream>>>(bufA, wd3, bd3, bufB);
  convt3x3_s2_act< 8,  1, 50, 50, 1, true ><<<dim3(40,  1, N), 256, 0, stream>>>(bufB, wd4, bd4, out);
}
